// GKT_24060406792370
// MI455X (gfx1250) — matrix operand order repaired and checked by reading, compiled, not run on hardware
//
#include <hip/hip_runtime.h>
#include <hip/hip_bf16.h>

// Problem constants (match reference)
#define Bsz 256
#define Tsz 64
#define Nsz 128
#define Hsz 128
#define BPW 16           // batch rows per workgroup (M dim of WMMA tiles)
#define NTHREADS 256     // 8 wave32 -> 8 column tiles of 16 = H columns

typedef __attribute__((ext_vector_type(16))) __bf16 v16bf;
typedef __attribute__((ext_vector_type(8)))  __bf16 v8bf;
typedef __attribute__((ext_vector_type(8)))  float  v8f;

// exact round-to-nearest-even float -> bf16
static __device__ __forceinline__ __bf16 f2bf(float f) {
  union { float f; unsigned u; } v; v.f = f;
  unsigned r = v.u + 0x7fffu + ((v.u >> 16) & 1u);
  unsigned short hs = (unsigned short)(r >> 16);
  __bf16 o; __builtin_memcpy(&o, &hs, 2); return o;
}

static __device__ __forceinline__ float sigmoidf_(float x) {
  return 1.0f / (1.0f + __expf(-x));
}

// A-matrix (16x32 bf16) fragment, contiguous k order (matches B):
// lanes 0-15 row=lane: elements hold K 0..15
// lanes 16-31: elements hold K 16..31
// => 16 contiguous elements at kbase+16*half
static __device__ __forceinline__ v16bf load_a(const __bf16* row, int kbase, int half) {
  const __bf16* p = row + kbase + 16 * half;
  v8bf lo = *(const v8bf*)(p);
  v8bf hi = *(const v8bf*)(p + 8);
  v16bf a;
#pragma unroll
  for (int i = 0; i < 8; ++i) { a[i] = lo[i]; a[8 + i] = hi[i]; }
  return a;
}

// One-time: fp32 weights -> bf16 copies in d_ws (row-major [n][k],
// which is exactly the B-matrix access order B[k][n] = W[n][k])
__global__ void gkt_prep(const float* __restrict__ wih, const float* __restrict__ whh,
                         __bf16* __restrict__ wihb, __bf16* __restrict__ whhb) {
  int i = blockIdx.x * 256 + threadIdx.x;
  if (i < 3 * Hsz * 2 * Hsz) wihb[i] = f2bf(wih[i]);
  if (i < 3 * Hsz * Hsz)     whhb[i] = f2bf(whh[i]);
}

__global__ __launch_bounds__(NTHREADS)
void gkt_main(const int* __restrict__ task_seq, const int* __restrict__ status_seq,
              const float* __restrict__ adj, const float* __restrict__ emb,
              const float* __restrict__ b_ih, const float* __restrict__ b_hh,
              const float* __restrict__ w_pred, const float* __restrict__ b_pred,
              const __bf16* __restrict__ wihb, const __bf16* __restrict__ whhb,
              float* __restrict__ logits, float* __restrict__ hstate) {
  __shared__ int   task_s[BPW], stat_s[BPW];
  __shared__ float adjs[BPW][Nsz];                        // adj rows for the 16 tasks
  __shared__ __align__(32) __bf16 xcat[BPW][2 * Hsz];     // concat(x, agg), WMMA A operand
  __shared__ __align__(32) __bf16 prevb[BPW][Hsz];        // prev hidden, WMMA A operand
  __shared__ float prevf[BPW][Hsz];                       // prev hidden fp32 for combine
  __shared__ float hnew[BPW][Hsz];                        // updated rows (for dot update)
  __shared__ float Dt[BPW][Nsz];                          // running h[b] @ w_pred table
  __shared__ float wps[Hsz];

  const int tid  = threadIdx.x;
  const int wave = tid >> 5;
  const int lane = tid & 31;
  const int half = lane >> 4;
  const int lrow = lane & 15;
  const int c    = wave * 16;                 // this wave's 16-column slice of H
  const int b_base = blockIdx.x * BPW;

  const float bpred = b_pred[0];
  const float bihr = b_ih[c + lrow], bihz = b_ih[Hsz + c + lrow], bihn = b_ih[2 * Hsz + c + lrow];
  const float bhhr = b_hh[c + lrow], bhhz = b_hh[Hsz + c + lrow], bhhn = b_hh[2 * Hsz + c + lrow];

  // h state lives in the h_final region of d_out; init this WG's slice to zero.
  {
    float* hp = hstate + (size_t)b_base * Nsz * Hsz;
    for (int i = tid; i < BPW * Nsz * Hsz; i += NTHREADS) hp[i] = 0.0f;
  }
  if (tid < Hsz) wps[tid] = w_pred[tid];
  for (int i = tid; i < BPW * Nsz; i += NTHREADS) (&Dt[0][0])[i] = 0.0f;
  __threadfence();
  __syncthreads();

  const int srow = tid >> 4;          // staging: row 0..15
  const int j0   = (tid & 15) * 8;    // staging: 8-column chunk

  for (int t = 0; t < Tsz; ++t) {
    if (tid < BPW) {
      task_s[tid] = task_seq[(b_base + tid) * Tsz + t];
      stat_s[tid] = status_seq[(b_base + tid) * Tsz + t];
    }
    __syncthreads();

    // ---- stage adj row, interaction embedding x, prev hidden row ----
    {
      const int tk = task_s[srow], st = stat_s[srow];
      const float* ar = adj + (size_t)tk * Nsz;
      const float* xr = emb + ((size_t)tk * 3 + st) * Hsz;
      const float* pr = hstate + ((size_t)(b_base + srow) * Nsz + tk) * Hsz;
#pragma unroll
      for (int j = 0; j < 8; ++j) {
        adjs[srow][j0 + j] = ar[j0 + j];
        xcat[srow][j0 + j] = f2bf(xr[j0 + j]);
        float pv = pr[j0 + j];
        prevf[srow][j0 + j] = pv;
        prevb[srow][j0 + j] = f2bf(pv);
      }
    }
    __syncthreads();

    // ---- agg = adj[task] @ h[b] (sparse-aware: adj rows are ~95% exact zeros) ----
    {
      float acc[8] = {0.f, 0.f, 0.f, 0.f, 0.f, 0.f, 0.f, 0.f};
      const float* hb = hstate + (size_t)(b_base + srow) * Nsz * Hsz + j0;
      for (int n = 0; n < Nsz; ++n) {
        float a = adjs[srow][n];
        if (a != 0.0f) {
          const float4* hv = (const float4*)(hb + (size_t)n * Hsz);
          float4 h0 = hv[0], h1 = hv[1];
          acc[0] += a * h0.x; acc[1] += a * h0.y; acc[2] += a * h0.z; acc[3] += a * h0.w;
          acc[4] += a * h1.x; acc[5] += a * h1.y; acc[6] += a * h1.z; acc[7] += a * h1.w;
        }
      }
#pragma unroll
      for (int j = 0; j < 8; ++j) xcat[srow][Hsz + j0 + j] = f2bf(acc[j]);
    }
    __syncthreads();

    // ---- GRU gate GEMMs via WMMA bf16 ----
    // wave w computes columns [c, c+16) of gi_r, gi_z, gi_n, gh_r, gh_z, gh_n
    v8f acc0 = {}, acc1 = {}, acc2 = {}, acc3 = {}, acc4 = {}, acc5 = {};
#pragma unroll
    for (int ks = 0; ks < 8; ++ks) {            // K = 256 over concat(x, agg)
      v16bf a = load_a(&xcat[lrow][0], ks * 32, half);
      // B[k][n] = W_ih[n][k]; fragment: 16 contiguous k at row n = g*128 + c + lrow
      const __bf16* wb = wihb + (size_t)(c + lrow) * (2 * Hsz) + ks * 32 + 16 * half;
      v16bf b0 = *(const v16bf*)(wb);
      v16bf b1 = *(const v16bf*)(wb + (size_t)Hsz * (2 * Hsz));
      v16bf b2 = *(const v16bf*)(wb + (size_t)(2 * Hsz) * (2 * Hsz));
      acc0 = __builtin_amdgcn_wmma_f32_16x16x32_bf16(false, a, false, b0, (short)0, acc0, false, false);
      acc1 = __builtin_amdgcn_wmma_f32_16x16x32_bf16(false, a, false, b1, (short)0, acc1, false, false);
      acc2 = __builtin_amdgcn_wmma_f32_16x16x32_bf16(false, a, false, b2, (short)0, acc2, false, false);
    }
#pragma unroll
    for (int ks = 0; ks < 4; ++ks) {            // K = 128 over prev hidden
      v16bf a = load_a(&prevb[lrow][0], ks * 32, half);
      const __bf16* wb = whhb + (size_t)(c + lrow) * Hsz + ks * 32 + 16 * half;
      v16bf b0 = *(const v16bf*)(wb);
      v16bf b1 = *(const v16bf*)(wb + (size_t)Hsz * Hsz);
      v16bf b2 = *(const v16bf*)(wb + (size_t)(2 * Hsz) * Hsz);
      acc3 = __builtin_amdgcn_wmma_f32_16x16x32_bf16(false, a, false, b0, (short)0, acc3, false, false);
      acc4 = __builtin_amdgcn_wmma_f32_16x16x32_bf16(false, a, false, b1, (short)0, acc4, false, false);
      acc5 = __builtin_amdgcn_wmma_f32_16x16x32_bf16(false, a, false, b2, (short)0, acc5, false, false);
    }

    // ---- gate math + scatter updated row into h ----
    // C/D layout: element v <-> (row = v + 8*half, col = lane&15)
#pragma unroll
    for (int v = 0; v < 8; ++v) {
      const int row = v + 8 * half;
      float r  = sigmoidf_(acc0[v] + acc3[v] + bihr + bhhr);
      float z  = sigmoidf_(acc1[v] + acc4[v] + bihz + bhhz);
      float nn = tanhf(acc2[v] + bihn + r * (acc5[v] + bhhn));
      float pv = prevf[row][c + lrow];
      float hv = (1.0f - z) * nn + z * pv;
      hnew[row][c + lrow] = hv;
      hstate[((size_t)(b_base + row) * Nsz + task_s[row]) * Hsz + c + lrow] = hv;
    }
    __threadfence();
    __syncthreads();

    // ---- incremental predict-layer: only the task row's dot changed ----
    if (tid < BPW) {
      float s = 0.0f;
      for (int j = 0; j < Hsz; ++j) s += hnew[tid][j] * wps[j];
      Dt[tid][task_s[tid]] = s;
    }
    __syncthreads();

    // ---- emit logits[b, t, :] = Dt row + bias ----
#pragma unroll
    for (int j = 0; j < 8; ++j) {
      logits[((size_t)(b_base + srow) * Tsz + t) * Nsz + j0 + j] = Dt[srow][j0 + j] + bpred;
    }
    __syncthreads();
  }
}

extern "C" void kernel_launch(void* const* d_in, const int* in_sizes, int n_in,
                              void* d_out, int out_size, void* d_ws, size_t ws_size,
                              hipStream_t stream) {
  (void)in_sizes; (void)n_in; (void)out_size; (void)ws_size;
  const int*   task_seq   = (const int*)d_in[0];
  const int*   status_seq = (const int*)d_in[1];
  const float* adj        = (const float*)d_in[2];
  const float* emb        = (const float*)d_in[3];
  const float* W_ih       = (const float*)d_in[4];
  const float* W_hh       = (const float*)d_in[5];
  const float* b_ih       = (const float*)d_in[6];
  const float* b_hh       = (const float*)d_in[7];
  const float* w_pred     = (const float*)d_in[8];
  const float* b_pred     = (const float*)d_in[9];

  float* logits = (float*)d_out;                              // [B,T,N]
  float* hstate = logits + (size_t)Bsz * Tsz * Nsz;           // [B,N,H] (h_final region)

  __bf16* wihb = (__bf16*)d_ws;                               // [3H][2H] bf16
  __bf16* whhb = wihb + (size_t)3 * Hsz * 2 * Hsz;            // [3H][H]  bf16

  gkt_prep<<<dim3((3 * Hsz * 2 * Hsz + 255) / 256), dim3(256), 0, stream>>>(W_ih, W_hh, wihb, whhb);
  gkt_main<<<dim3(Bsz / BPW), dim3(NTHREADS), 0, stream>>>(
      task_seq, status_seq, adj, emb, b_ih, b_hh, w_pred, b_pred,
      wihb, whhb, logits, hstate);
}
